// EdgeModel_25598005084721
// MI455X (gfx1250) — matrix operand order repaired and checked by reading, not compiled, not run on hardware
//
#include <hip/hip_runtime.h>

typedef __attribute__((ext_vector_type(16))) __bf16 v16bf;
typedef __attribute__((ext_vector_type(8)))  __bf16 v8bf;
typedef __attribute__((ext_vector_type(8)))  float  v8f;
typedef __attribute__((ext_vector_type(4)))  float  v4f;

// f32 -> bf16 native convert (RNE); lowers to v_cvt_pk_bf16_f32 pairs
static __device__ __forceinline__ __bf16 f2bf(float f) {
    return (__bf16)f;
}

// ---------------------------------------------------------------------------
// Prep: convert W1..W4 (f32, [K][N] row-major) -> bf16 transposed [N][K] in ws.
//   W1t: [128][512] at elem offset 0
//   W2t: [128][128] at 65536 ; W3t at 81920 ; W4t at 98304  (total 224 KB)
// ---------------------------------------------------------------------------
__global__ void prep_weights(const float* __restrict__ W1,
                             const float* __restrict__ W2,
                             const float* __restrict__ W3,
                             const float* __restrict__ W4,
                             __bf16* __restrict__ ws) {
    int i = blockIdx.x * 256 + threadIdx.x;
    if (i < 65536) {                       // W1t [n][k], K=512
        int n = i >> 9, k = i & 511;
        ws[i] = f2bf(W1[k * 128 + n]);
    } else if (i < 81920) {                // W2t
        int j = i - 65536; int n = j >> 7, k = j & 127;
        ws[i] = f2bf(W2[k * 128 + n]);
    } else if (i < 98304) {                // W3t
        int j = i - 81920; int n = j >> 7, k = j & 127;
        ws[i] = f2bf(W3[k * 128 + n]);
    } else if (i < 114688) {               // W4t
        int j = i - 98304; int n = j >> 7, k = j & 127;
        ws[i] = f2bf(W4[k * 128 + n]);
    }
}

// ---------------------------------------------------------------------------
// Fused MLP. Block = 256 threads = 8 waves; each wave owns TWO 16-row strips
// (M=32 register tile) -> every B fragment is reused by 2 wmmas, halving
// B-side cache traffic per FLOP. Block covers 256 edges; grid = E/256 = 1280.
// Streaming data (X, out) is non-temporal; weights stay cache-hot.
// ---------------------------------------------------------------------------
__launch_bounds__(256)
__global__ void edge_mlp_kernel(const float* __restrict__ recv,
                                const float* __restrict__ send,
                                const float* __restrict__ eatt,
                                const float* __restrict__ gatt,
                                const __bf16* __restrict__ wt,
                                const float* __restrict__ b1,
                                const float* __restrict__ b2,
                                const float* __restrict__ b3,
                                const float* __restrict__ b4,
                                float* __restrict__ out) {
    __shared__ __bf16 Hs[256 * 128];     // 64 KB inter-layer activation buffer

    const int lane = threadIdx.x & 31;
    const int wave = threadIdx.x >> 5;   // 0..7
    const int mr   = lane & 15;          // row within strip / col within n-tile
    const int h    = lane >> 4;          // half-wave select

    // Two edge rows per lane (strip 0 and strip 1)
    const long e0 = (long)blockIdx.x * 256 + wave * 32 + mr;   // strip 0
    const long e1 = e0 + 16;                                   // strip 1

    const __bf16* W1t = wt;              // [128][512]
    const __bf16* W2t = wt + 65536;      // [128][128]
    const __bf16* W3t = wt + 81920;
    const __bf16* W4t = wt + 98304;

    const float* rowr_[2] = { recv + e0 * 128, recv + e1 * 128 };
    const float* rows_[2] = { send + e0 * 128, send + e1 * 128 };
    const float* rowe_[2] = { eatt + e0 * 128, eatt + e1 * 128 };

    v8f acc[2][8];                        // [strip][n-tile]
    const v8f vzero = {};

    // Build layer-1 A fragment for strip s, 32-wide K slab ks.
    // A 16x32 fragment, this code's contiguous k order (matches B): lane holds K = [h*16 .. +15].
    auto loadA1 = [&](int ks, int s, v16bf& A) {
        const int kb = ks * 32;
        const float* src;
        bool stream = true;
        if (kb < 128)       { src = rowr_[s] + kb; }
        else if (kb < 256)  { src = rows_[s] + (kb - 128); }
        else if (kb < 384)  { src = rowe_[s] + (kb - 256); }
        else                { src = gatt + (kb - 384); stream = false; } // broadcast row

        const float* p0 = src + h * 16;
        const float* p1 = src + h * 16 + 8;
        v4f f0, f1, f2, f3;
        if (stream) {
            f0 = __builtin_nontemporal_load((const v4f*)(p0));
            f1 = __builtin_nontemporal_load((const v4f*)(p0 + 4));
            f2 = __builtin_nontemporal_load((const v4f*)(p1));
            f3 = __builtin_nontemporal_load((const v4f*)(p1 + 4));
        } else {
            f0 = *(const v4f*)(p0);
            f1 = *(const v4f*)(p0 + 4);
            f2 = *(const v4f*)(p1);
            f3 = *(const v4f*)(p1 + 4);
        }
#pragma unroll
        for (int i = 0; i < 4; ++i) {
            A[i]      = f2bf(f0[i]);
            A[4 + i]  = f2bf(f1[i]);
            A[8 + i]  = f2bf(f2[i]);
            A[12 + i] = f2bf(f3[i]);
        }
    };

    // Bias + relu + store acc to LDS in C/D layout (lane: col mr, rows r+8h).
    auto storeH = [&](const float* bias) {
#pragma unroll
        for (int s = 0; s < 2; ++s) {
#pragma unroll
            for (int nt = 0; nt < 8; ++nt) {
                float bv = bias[nt * 16 + mr];
#pragma unroll
                for (int r = 0; r < 8; ++r) {
                    float v = fmaxf(acc[s][nt][r] + bv, 0.0f);
                    Hs[(wave * 32 + s * 16 + r + 8 * h) * 128 + nt * 16 + mr] = f2bf(v);
                }
            }
        }
    };

    // ---------------- Layer 1: X[256x512] @ W1 -> acc ----------------
#pragma unroll
    for (int s = 0; s < 2; ++s)
#pragma unroll
        for (int nt = 0; nt < 8; ++nt) acc[s][nt] = vzero;

    v16bf Abuf[2][2];                     // [buf parity][strip]
    loadA1(0, 0, Abuf[0][0]);
    loadA1(0, 1, Abuf[0][1]);

#pragma unroll
    for (int ks = 0; ks < 16; ++ks) {
        const int kb  = ks * 32;
        const int cur = ks & 1, nxt = cur ^ 1;

        // Prefetch + convert next A pair into alternate buffers (overlaps the
        // wmma chain; no WMMA->VALU WAR on live A regs).
        if (ks < 15) {
            loadA1(ks + 1, 0, Abuf[nxt][0]);
            loadA1(ks + 1, 1, Abuf[nxt][1]);
        }

        // B in two half-batches of 4 n-tiles to bound register pressure.
        // B 32x16 fragment, same contiguous k order as A: lane holds col n = nt*16+mr, K = [h*16 .. +15].
#pragma unroll
        for (int half = 0; half < 2; ++half) {
            v16bf Bf[4];
#pragma unroll
            for (int j = 0; j < 4; ++j) {
                const int nt = half * 4 + j;
                Bf[j] = *(const v16bf*)(W1t + (nt * 16 + mr) * 512 + kb + h * 16);
            }
#pragma unroll
            for (int j = 0; j < 4; ++j) {
                const int nt = half * 4 + j;
                acc[0][nt] = __builtin_amdgcn_wmma_f32_16x16x32_bf16(
                    false, Abuf[cur][0], false, Bf[j], (short)0, acc[0][nt], false, false);
                acc[1][nt] = __builtin_amdgcn_wmma_f32_16x16x32_bf16(
                    false, Abuf[cur][1], false, Bf[j], (short)0, acc[1][nt], false, false);
            }
        }
    }

    storeH(b1);
    __syncthreads();

    // ---------------- Layers 2 & 3: H[256x128] @ W -> H ----------------
    const __bf16* hrow0 = Hs + (wave * 32 + mr) * 128;
    const __bf16* hrow1 = Hs + (wave * 32 + 16 + mr) * 128;

    const __bf16* Wmid[2] = {W2t, W3t};
    const float*  bmid[2] = {b2, b3};
    for (int L = 0; L < 2; ++L) {
#pragma unroll
        for (int s = 0; s < 2; ++s)
#pragma unroll
            for (int nt = 0; nt < 8; ++nt) acc[s][nt] = vzero;

#pragma unroll
        for (int ks = 0; ks < 4; ++ks) {
            const int kb = ks * 32;
            union { v16bf v; v8bf half2[2]; } a0, a1;
            a0.half2[0] = *(const v8bf*)(hrow0 + kb + h * 16);
            a0.half2[1] = *(const v8bf*)(hrow0 + kb + h * 16 + 8);
            a1.half2[0] = *(const v8bf*)(hrow1 + kb + h * 16);
            a1.half2[1] = *(const v8bf*)(hrow1 + kb + h * 16 + 8);
#pragma unroll
            for (int half = 0; half < 2; ++half) {
                v16bf Bf[4];
#pragma unroll
                for (int j = 0; j < 4; ++j) {
                    const int nt = half * 4 + j;
                    Bf[j] = *(const v16bf*)(Wmid[L] + (nt * 16 + mr) * 128 + kb + h * 16);
                }
#pragma unroll
                for (int j = 0; j < 4; ++j) {
                    const int nt = half * 4 + j;
                    acc[0][nt] = __builtin_amdgcn_wmma_f32_16x16x32_bf16(
                        false, a0.v, false, Bf[j], (short)0, acc[0][nt], false, false);
                    acc[1][nt] = __builtin_amdgcn_wmma_f32_16x16x32_bf16(
                        false, a1.v, false, Bf[j], (short)0, acc[1][nt], false, false);
                }
            }
        }
        __syncthreads();   // all strip reads done before overwrite
        storeH(bmid[L]);
        __syncthreads();
    }

    // ---------------- Layer 4: H @ W4 + b4 -> out (f32, no relu) ----------------
#pragma unroll
    for (int s = 0; s < 2; ++s)
#pragma unroll
        for (int nt = 0; nt < 8; ++nt) acc[s][nt] = vzero;

#pragma unroll
    for (int ks = 0; ks < 4; ++ks) {
        const int kb = ks * 32;
        union { v16bf v; v8bf half2[2]; } a0, a1;
        a0.half2[0] = *(const v8bf*)(hrow0 + kb + h * 16);
        a0.half2[1] = *(const v8bf*)(hrow0 + kb + h * 16 + 8);
        a1.half2[0] = *(const v8bf*)(hrow1 + kb + h * 16);
        a1.half2[1] = *(const v8bf*)(hrow1 + kb + h * 16 + 8);
#pragma unroll
        for (int half = 0; half < 2; ++half) {
            v16bf Bf[4];
#pragma unroll
            for (int j = 0; j < 4; ++j) {
                const int nt = half * 4 + j;
                Bf[j] = *(const v16bf*)(W4t + (nt * 16 + mr) * 128 + kb + h * 16);
            }
#pragma unroll
            for (int j = 0; j < 4; ++j) {
                const int nt = half * 4 + j;
                acc[0][nt] = __builtin_amdgcn_wmma_f32_16x16x32_bf16(
                    false, a0.v, false, Bf[j], (short)0, acc[0][nt], false, false);
                acc[1][nt] = __builtin_amdgcn_wmma_f32_16x16x32_bf16(
                    false, a1.v, false, Bf[j], (short)0, acc[1][nt], false, false);
            }
        }
    }

#pragma unroll
    for (int s = 0; s < 2; ++s) {
#pragma unroll
        for (int nt = 0; nt < 8; ++nt) {
            float bv = b4[nt * 16 + mr];
#pragma unroll
            for (int r = 0; r < 8; ++r) {
                long row = (long)blockIdx.x * 256 + wave * 32 + s * 16 + r + 8 * h;
                __builtin_nontemporal_store(acc[s][nt][r] + bv,
                                            out + row * 128 + nt * 16 + mr);
            }
        }
    }
}

// ---------------------------------------------------------------------------
extern "C" void kernel_launch(void* const* d_in, const int* in_sizes, int n_in,
                              void* d_out, int out_size, void* d_ws, size_t ws_size,
                              hipStream_t stream) {
    const float* recv = (const float*)d_in[0];
    const float* send = (const float*)d_in[1];
    const float* eatt = (const float*)d_in[2];
    const float* gatt = (const float*)d_in[3];
    const float* W1   = (const float*)d_in[4];
    const float* b1   = (const float*)d_in[5];
    const float* W2   = (const float*)d_in[6];
    const float* b2   = (const float*)d_in[7];
    const float* W3   = (const float*)d_in[8];
    const float* b3   = (const float*)d_in[9];
    const float* W4   = (const float*)d_in[10];
    const float* b4   = (const float*)d_in[11];
    __bf16* wt = (__bf16*)d_ws;   // needs 229,376 bytes of workspace

    prep_weights<<<448, 256, 0, stream>>>(W1, W2, W3, W4, wt);

    // E = 327680 edges, 256 edges per block
    edge_mlp_kernel<<<1280, 256, 0, stream>>>(recv, send, eatt, gatt, wt,
                                              b1, b2, b3, b4, (float*)d_out);
}
